// WOP_26121991094405
// MI455X (gfx1250) — matrix operand order repaired and checked by reading, not compiled, not run on hardware
//
#include <hip/hip_runtime.h>
#include <hip/hip_bf16.h>
#include <math.h>

typedef __attribute__((ext_vector_type(16))) __bf16   v16bf;
typedef __attribute__((ext_vector_type(8)))  float    v8f;
typedef __attribute__((ext_vector_type(4)))  unsigned u32x4;
typedef __attribute__((ext_vector_type(4)))  float    f32x4;

#define BATCH 16384
#define LW    4
#define IS    1024
#define HS    512
#define NOPS  3

#define M_TILE     64
#define K_CHUNK    128
#define NCHUNK     (IS / K_CHUNK)             /* 8 */
#define X_STRIDE_B ((K_CHUNK + 8) * 2)        /* 272 B, padded row of bf16 */
#define W_STRIDE_B X_STRIDE_B
#define X_BYTES    (M_TILE * X_STRIDE_B)      /* 17408  */
#define W_BYTES    (HS * W_STRIDE_B)          /* 139264 */
#define SMEM_BYTES (2 * X_BYTES + 2 * W_BYTES) /* 313344 <= 320K LDS */
#define H_STRIDE_B ((HS + 4) * 4)             /* 2064 B, padded fp32 row */

union Frag { u32x4 q[2]; v16bf v; };

__device__ __forceinline__ unsigned short f2bf(float f) {
    unsigned u = __float_as_uint(f);
    u += 0x7fffu + ((u >> 16) & 1u);          /* round-to-nearest-even */
    return (unsigned short)(u >> 16);
}
__device__ __forceinline__ unsigned pack2(float a, float b) {
    return (unsigned)f2bf(a) | ((unsigned)f2bf(b) << 16);
}

/* async global -> LDS copy, 16 B per lane; tracked by ASYNCcnt */
__device__ __forceinline__ void async_copy16(unsigned lds_off, unsigned long long gaddr) {
    asm volatile("global_load_async_to_lds_b128 %0, %1, off"
                 :: "v"(lds_off), "v"(gaddr) : "memory");
}
__device__ __forceinline__ void wait_async0() {
#if __has_builtin(__builtin_amdgcn_s_wait_asynccnt)
    __builtin_amdgcn_s_wait_asynccnt(0);
#else
    asm volatile("s_wait_asynccnt 0" ::: "memory");
#endif
}

/* ---------------- prep: W1 fp32 [IS][HS] -> bf16 transposed [HS][IS] ------ */
__global__ void __launch_bounds__(256)
wop_prep_w1t(const float* __restrict__ W1, unsigned short* __restrict__ w1t) {
    int idx = blockIdx.x * 256 + threadIdx.x;     /* idx = k*HS + n */
    if (idx < IS * HS) {
        int k = idx >> 9;          /* HS = 512 */
        int n = idx & (HS - 1);
        w1t[n * IS + k] = f2bf(W1[idx]);
    }
}

/* ---- helpers for the pipelined main loop -------------------------------- */
__device__ __forceinline__ void issue_w1t_async(char* wbuf, const unsigned short* w1t,
                                                int k0, int t) {
    unsigned long long gbase = (unsigned long long)(const char*)w1t + (size_t)k0 * 2;
    #pragma unroll 4
    for (int i = 0; i < 32; ++i) {
        int f   = i * 256 + t;
        int n   = f >> 4;        /* W1T row             */
        int seg = f & 15;        /* 16-B segment in row */
        async_copy16((unsigned)(size_t)(wbuf + n * W_STRIDE_B + seg * 16),
                     gbase + (unsigned long long)n * (IS * 2) + seg * 16);
    }
}

__device__ __forceinline__ void issue_x_loads(f32x4* xv, f32x4* rv,
                                              const float* wenc, const float* repeat,
                                              int m0, int k0, int wave, int lane,
                                              unsigned repPack) {
    #pragma unroll
    for (int i = 0; i < 8; ++i) {
        int Ri = m0 + 8 * i + wave;
        xv[i] = *(const f32x4*)(wenc + (size_t)Ri * IS + k0 + lane * 4);
        int rp = (repPack >> (2 * i)) & 3;
        rv[i] = *(const f32x4*)(repeat + rp * IS + k0 + lane * 4);
    }
}

__device__ __forceinline__ void pack_store_x(char* xbuf, const f32x4* xv, const f32x4* rv,
                                             int wave, int lane,
                                             unsigned validMask, unsigned repMask) {
    #pragma unroll
    for (int i = 0; i < 8; ++i) {
        f32x4 v = xv[i];
        if (repMask & (1u << i)) v += rv[i];
        if (!(validMask & (1u << i))) { v.x = 0.f; v.y = 0.f; v.z = 0.f; v.w = 0.f; }
        unsigned lo = pack2(v.x, v.y), hi = pack2(v.z, v.w);
        unsigned long long p = (unsigned long long)lo | ((unsigned long long)hi << 32);
        *(unsigned long long*)(xbuf + (8 * i + wave) * X_STRIDE_B + lane * 8) = p;
    }
}

__device__ __forceinline__ Frag loadB(const char* wbuf, int n, int kk, int lane) {
    Frag B;
    const char* p = wbuf + n * W_STRIDE_B + (kk + 8 * (lane >> 4)) * 2;
    B.q[0] = *(const u32x4*)p;               /* K = kk+8h .. +7     */
    B.q[1] = *(const u32x4*)(p + 32);        /* K = kk+16+8h .. +7  */
    return B;
}

/* ---------------- main fused kernel -------------------------------------- */
__global__ void __launch_bounds__(256)
wop_main(const float* __restrict__ wenc, const float* __restrict__ repeat,
         const float* __restrict__ b1,   const float* __restrict__ W2,
         const float* __restrict__ b2,   const int*   __restrict__ wn,
         const int*   __restrict__ wc,   const unsigned short* __restrict__ w1t,
         float* __restrict__ out)
{
    extern __shared__ char smem[];

    const int t    = threadIdx.x;
    const int lane = t & 31;
    const int wave = t >> 5;
    const int m0   = blockIdx.x * M_TILE;

    /* ---- per-row metadata for the 8 rows this thread's wave stages ------ */
    unsigned validMask = 0, repMask = 0, repPack = 0;
    #pragma unroll
    for (int i = 0; i < 8; ++i) {
        int Ri = m0 + 8 * i + wave;
        int b  = Ri >> 2, l = Ri & 3;
        int c0 = wc[4 * b], c1 = wc[4 * b + 1], c2 = wc[4 * b + 2], c3 = wc[4 * b + 3];
        int cl = (l == 0) ? c0 : (l == 1) ? c1 : (l == 2) ? c2 : c3;
        int rep = (int)(l > 0 && c0 == cl) + (int)(l > 1 && c1 == cl) + (int)(l > 2 && c2 == cl);
        bool valid = l < wn[b];
        if (valid)            validMask |= 1u << i;
        if (valid && rep > 0) repMask   |= 1u << i;
        repPack |= ((unsigned)rep) << (2 * i);
    }

    const int nBase = (wave & 3) * 128;   /* 4 waves across N = 512 */
    const int mBase = (wave >> 2) * 32;   /* 2 waves across M = 64  */

    v8f acc[2][8];
    #pragma unroll
    for (int i = 0; i < 2; ++i)
        #pragma unroll
        for (int j = 0; j < 8; ++j)
            #pragma unroll
            for (int e = 0; e < 8; ++e) acc[i][j][e] = 0.f;

    f32x4 xv[8], rv[8];

    /* ---- prologue: fully stage chunk 0 into buffer 0 -------------------- */
    issue_w1t_async(smem + 2 * X_BYTES, w1t, 0, t);
    issue_x_loads(xv, rv, wenc, repeat, m0, 0, wave, lane, repPack);
    pack_store_x(smem, xv, rv, wave, lane, validMask, repMask);

    /* ---- software-pipelined K loop: stage c+1 under compute of c -------- */
    for (int c = 0; c < NCHUNK; ++c) {
        const int pb = c & 1;
        char* xcur = smem + pb * X_BYTES;
        char* wcur = smem + 2 * X_BYTES + pb * W_BYTES;
        char* xnxt = smem + (pb ^ 1) * X_BYTES;
        char* wnxt = smem + 2 * X_BYTES + (pb ^ 1) * W_BYTES;

        wait_async0();          /* our async writes for chunk c are done    */
        __syncthreads();        /* everyone's staging of chunk c is visible */

        const bool have_next = (c + 1 < NCHUNK);
        if (have_next) {
            issue_w1t_async(wnxt, w1t, (c + 1) * K_CHUNK, t);   /* ASYNCcnt */
            issue_x_loads(xv, rv, wenc, repeat, m0, (c + 1) * K_CHUNK,
                          wave, lane, repPack);                  /* LOADcnt */
        }

        /* compute on chunk c: 64 WMMAs/wave, B-fragment register pipeline */
        #pragma unroll
        for (int kk = 0; kk < K_CHUNK; kk += 32) {
            Frag A[2];
            #pragma unroll
            for (int ms = 0; ms < 2; ++ms) {
                const char* p = xcur + (mBase + 16 * ms + (lane & 15)) * X_STRIDE_B
                                     + (kk + 8 * (lane >> 4)) * 2;
                A[ms].q[0] = *(const u32x4*)p;           /* K = kk+8h .. +7  */
                A[ms].q[1] = *(const u32x4*)(p + 32);    /* K = kk+16+8h ..  */
            }
            Frag Bc = loadB(wcur, nBase, kk, lane);
            #pragma unroll
            for (int ns = 0; ns < 8; ++ns) {
                Frag Bn;
                if (ns < 7) Bn = loadB(wcur, nBase + 16 * (ns + 1), kk, lane);
                acc[0][ns] = __builtin_amdgcn_wmma_f32_16x16x32_bf16(
                    false, A[0].v, false, Bc.v, (short)0, acc[0][ns], false, false);
                acc[1][ns] = __builtin_amdgcn_wmma_f32_16x16x32_bf16(
                    false, A[1].v, false, Bc.v, (short)0, acc[1][ns], false, false);
                if (ns < 7) Bc = Bn;
            }
        }

        /* pack + LDS-store next x chunk after WMMAs (hides HBM latency)    */
        if (have_next)
            pack_store_x(xnxt, xv, rv, wave, lane, validMask, repMask);
    }
    __syncthreads();            /* all compute done before smem is reused   */

    /* ---- epilogue 1: h = tanh(acc + b1) -> LDS fp32 [64][HS+pad] -------- */
    #pragma unroll
    for (int ns = 0; ns < 8; ++ns) {
        const int   n    = nBase + 16 * ns + (lane & 15);
        const float bias = b1[n];
        #pragma unroll
        for (int ms = 0; ms < 2; ++ms) {
            const int mrow = mBase + 16 * ms + 8 * (lane >> 4);
            #pragma unroll
            for (int j = 0; j < 8; ++j) {
                float hv = tanhf(acc[ms][ns][j] + bias);
                *(float*)(smem + (mrow + j) * H_STRIDE_B + n * 4) = hv;
            }
        }
    }
    __syncthreads();

    /* ---- epilogue 2: s = h @ W2 + b2 (N = 3, VALU dot products) --------- */
    if (t < M_TILE * NOPS) {
        const int r = t / NOPS;
        const int o = t - r * NOPS;
        const char* hrow = smem + r * H_STRIDE_B;
        float s = 0.f;
        for (int i = 0; i < HS / 4; ++i) {
            f32x4 hv = *(const f32x4*)(hrow + i * 16);
            const float* wp = W2 + (i * 4) * NOPS + o;
            s += hv.x * wp[0] + hv.y * wp[NOPS] + hv.z * wp[2 * NOPS] + hv.w * wp[3 * NOPS];
        }
        out[(size_t)(m0 + r) * NOPS + o] = s + b2[o];
    }
}

extern "C" void kernel_launch(void* const* d_in, const int* in_sizes, int n_in,
                              void* d_out, int out_size, void* d_ws, size_t ws_size,
                              hipStream_t stream) {
    const float* wenc   = (const float*)d_in[0];
    const float* repeat = (const float*)d_in[1];
    const float* W1     = (const float*)d_in[2];
    const float* b1     = (const float*)d_in[3];
    const float* W2     = (const float*)d_in[4];
    const float* b2     = (const float*)d_in[5];
    const int*   wn     = (const int*)d_in[6];
    const int*   wc     = (const int*)d_in[7];
    float* out = (float*)d_out;
    unsigned short* w1t = (unsigned short*)d_ws;   /* 1 MB bf16 W1^T */

    wop_prep_w1t<<<(IS * HS + 255) / 256, 256, 0, stream>>>(W1, w1t);
    wop_main<<<(BATCH * LW) / M_TILE, 256, SMEM_BYTES, stream>>>(
        wenc, repeat, b1, W2, b2, wn, wc, w1t, out);
}
